// DeepseekMoEGate_44418551775973
// MI455X (gfx1250) — matrix operand order repaired and checked by reading, not compiled, not run on hardware
//
#include <hip/hip_runtime.h>
#include <hip/hip_bf16.h>

// DeepSeek MoE gate for gfx1250 (MI455X).
//   logits = X[T,4096] @ W^T[4096,256]   (bf16x3 split precision on WMMA)
//   topk_weight = softmax(top8 logits) * 2.5   (full softmax denom cancels)
// Outputs: d_out[0 .. T*8)        = top-k expert indices (as float)
//          d_out[T*8 .. 2*T*8)    = top-k normalized weights (float)

typedef __attribute__((ext_vector_type(16))) __bf16 v16bf;
typedef __attribute__((ext_vector_type(8)))  float  v8f;
typedef __attribute__((ext_vector_type(4)))  int    v4i;

#define H        4096
#define E        256
#define KCHUNK   32
#define NKC      (H / KCHUNK)     // 128 K-chunks
#define NTILE    (E / 16)         // 16 expert tiles
#define MBLOCK   128              // tokens per workgroup
#define TOPK     8
#define FRAG_DW  256              // dwords per B fragment (32 lanes x 8 dwords)
#define CHUNK_DW 4096             // dwords per (hi or lo) staged chunk

static __device__ __forceinline__ __bf16 f2bf(float f) { return (__bf16)f; }

static __device__ __forceinline__ unsigned pack2(__bf16 a, __bf16 b) {
    unsigned short sa, sb;
    __builtin_memcpy(&sa, &a, 2);
    __builtin_memcpy(&sb, &b, 2);
    return (unsigned)sa | ((unsigned)sb << 16);
}

// ---- CDNA5 async global->LDS path (ASYNCcnt), guarded for toolchain ------
#if __has_builtin(__builtin_amdgcn_global_load_async_to_lds_b128)
#define HAVE_ASYNC_LDS 1
typedef __attribute__((address_space(1))) v4i* gv4i_p;  // global
typedef __attribute__((address_space(3))) v4i* lv4i_p;  // LDS
template <int OFF>
static __device__ __forceinline__ void async_g2l_b128(const void* g, void* l) {
    // global_load_async_to_lds_b128: per-lane 16B, LDS written directly.
    __builtin_amdgcn_global_load_async_to_lds_b128(
        (gv4i_p)(g), (lv4i_p)(l), OFF, 0);
}
static __device__ __forceinline__ void wait_async0() {
#if __has_builtin(__builtin_amdgcn_s_wait_asynccnt)
    __builtin_amdgcn_s_wait_asynccnt(0);
#else
    asm volatile("s_wait_asynccnt 0x0" ::: "memory");
#endif
}
#else
#define HAVE_ASYNC_LDS 0
#endif

// ---------------------------------------------------------------------------
// Kernel 1: convert gate weight fp32[E,H] -> bf16 hi/lo fragment buffers,
// pre-swizzled into this code's wave32 B fragment (32x16, K x N) order, contiguous K:
//   fragment(kc, et): dword[lane*8 + j] holds (K = kc*32 + 2j + (lane>=16)*16,
//                                              K+1) for column N = lane%16.
// ---------------------------------------------------------------------------
__global__ void pack_w_kernel(const float* __restrict__ W,
                              unsigned* __restrict__ wsHi,
                              unsigned* __restrict__ wsLo) {
    int gid    = blockIdx.x * 256 + threadIdx.x;   // one dword of each buffer
    int frag   = gid >> 8;                         // 0 .. NKC*NTILE-1
    int within = gid & 255;
    int lane   = within >> 3;                      // 0..31
    int j      = within & 7;                       // 0..7
    int kc     = frag >> 4;
    int et     = frag & 15;
    int e      = et * 16 + (lane & 15);
    int k      = kc * KCHUNK + 2 * j + ((lane >> 4) << 4);

    float f0 = W[(size_t)e * H + k];
    float f1 = W[(size_t)e * H + k + 1];
    __bf16 h0 = f2bf(f0), h1 = f2bf(f1);
    __bf16 l0 = f2bf(f0 - (float)h0), l1 = f2bf(f1 - (float)h1);
    wsHi[gid] = pack2(h0, h1);
    wsLo[gid] = pack2(l0, l1);
}

// ---------------------------------------------------------------------------
// Kernel 2: fused GEMM + top-8 + renormalize.
// 8 waves, wave w owns token rows [w*16, w*16+16), all 256 experts.
// ---------------------------------------------------------------------------
__global__ void __launch_bounds__(256)
gate_kernel(const float* __restrict__ X,
            const uint4* __restrict__ BHi,   // packed fragments, hi
            const uint4* __restrict__ BLo,   // packed fragments, lo
            float* __restrict__ outIdx,
            float* __restrict__ outW) {
    // 128 KB LDS. During GEMM (async path): two 32 KB stage buffers
    // (dwords [0,8192) and [8192,16384)), each holding {hi:4096, lo:4096}.
    // Afterwards the whole region is reused for the 128x256 fp32 logits.
    __shared__ float smemf[MBLOCK * E];
    unsigned* smemu = (unsigned*)smemf;

    const int tid    = threadIdx.x;
    const int wv     = tid >> 5;          // wave 0..7
    const int lane   = tid & 31;
    const int rloc   = lane & 15;         // row within wave's 16-row strip
    const int klane  = lane >> 4;         // 0 or 1: A-layout K-offset group
    const int block0 = blockIdx.x * MBLOCK;

    v8f acc[NTILE] = {};                  // 16 tiles x 8 VGPRs fp32

    // A-matrix 16-bit layout (matches packed B): lanes 0-15 hold K {0..15},
    // lanes 16-31 hold K {16..31} of the same 16 rows.
    const float* xrow =
        X + (size_t)(block0 + wv * 16 + rloc) * H + klane * 16;

#if HAVE_ASYNC_LDS
    // Each thread copies 16B x 4 (hi) + 16B x 4 (lo) per chunk via the
    // direct-to-LDS async path; 8 waves x 8 instr x 512B = 32 KB per chunk.
    auto stage_async = [&](int kc, int buf) {
        const char* gH = (const char*)BHi + (size_t)kc * 16384 + tid * 16;
        const char* gL = (const char*)BLo + (size_t)kc * 16384 + tid * 16;
        char* lH = (char*)(smemu + buf * 2 * CHUNK_DW) + tid * 16;
        char* lL = lH + CHUNK_DW * 4;  // lo half of this stage buffer
        async_g2l_b128<0>(gH, lH);
        async_g2l_b128<4096>(gH, lH);
        async_g2l_b128<8192>(gH, lH);
        async_g2l_b128<12288>(gH, lH);
        async_g2l_b128<0>(gL, lL);
        async_g2l_b128<4096>(gL, lL);
        async_g2l_b128<8192>(gL, lL);
        async_g2l_b128<12288>(gL, lL);
    };
    stage_async(0, 0);
    wait_async0();
    __syncthreads();
#endif

    for (int kc = 0; kc < NKC; ++kc) {
#if HAVE_ASYNC_LDS
        const int cur = kc & 1;
        if (kc + 1 < NKC) stage_async(kc + 1, cur ^ 1);  // prefetch next chunk
        const unsigned bufbase = cur * 2 * CHUNK_DW;
#else
        __syncthreads();  // previous chunk's LDS reads done
        {   // Fallback: stage 32 KB of packed B through VGPRs.
            const uint4* srcH = BHi + (size_t)kc * 1024;
            const uint4* srcL = BLo + (size_t)kc * 1024;
            uint4* dstH = (uint4*)smemu;
            uint4* dstL = dstH + 1024;
#pragma unroll
            for (int i = 0; i < 4; ++i) {
                dstH[tid + i * 256] = srcH[tid + i * 256];
                dstL[tid + i * 256] = srcL[tid + i * 256];
            }
        }
        __syncthreads();
        const unsigned bufbase = 0;
#endif

        // Build A hi/lo fragments straight from global fp32 X.
        const float* xp = xrow + kc * KCHUNK;
        float4 p0 = *(const float4*)(xp + 0);
        float4 p1 = *(const float4*)(xp + 4);
        float4 p2 = *(const float4*)(xp + 8);
        float4 p3 = *(const float4*)(xp + 12);
        float fa[16] = {p0.x, p0.y, p0.z, p0.w, p1.x, p1.y, p1.z, p1.w,
                        p2.x, p2.y, p2.z, p2.w, p3.x, p3.y, p3.z, p3.w};
        v16bf ahi, alo;
#pragma unroll
        for (int i = 0; i < 16; ++i) {
            __bf16 h = f2bf(fa[i]);
            ahi[i] = h;
            alo[i] = f2bf(fa[i] - (float)h);
        }

        if (kc + 1 < NKC)  // pull next A chunk toward the caches
            __builtin_prefetch(xp + KCHUNK, 0, 1);

        // 16 expert tiles x (hi*hi + hi*lo + lo*hi)
#pragma unroll
        for (int t = 0; t < NTILE; ++t) {
            union { uint4 u[2]; v16bf v; } bh, bl;
            const uint4* bp =
                (const uint4*)(smemu + bufbase + t * FRAG_DW + lane * 8);
            const uint4* bq = (const uint4*)(smemu + bufbase + CHUNK_DW +
                                             t * FRAG_DW + lane * 8);
            bh.u[0] = bp[0]; bh.u[1] = bp[1];
            bl.u[0] = bq[0]; bl.u[1] = bq[1];
            acc[t] = __builtin_amdgcn_wmma_f32_16x16x32_bf16(
                false, ahi, false, bh.v, (short)0, acc[t], false, false);
            acc[t] = __builtin_amdgcn_wmma_f32_16x16x32_bf16(
                false, ahi, false, bl.v, (short)0, acc[t], false, false);
            acc[t] = __builtin_amdgcn_wmma_f32_16x16x32_bf16(
                false, alo, false, bh.v, (short)0, acc[t], false, false);
        }

#if HAVE_ASYNC_LDS
        wait_async0();    // own prefetch (chunk kc+1) landed in LDS
        __syncthreads();  // ...and everyone else's too
#endif
    }

    __syncthreads();  // done with B staging region; reuse LDS for logits

    // C/D layout: VGPR g -> row M=g (lanes 0-15) / M=g+8 (lanes 16-31), N=lane%16.
#pragma unroll
    for (int t = 0; t < NTILE; ++t)
#pragma unroll
        for (int g = 0; g < 8; ++g)
            smemf[(wv * 16 + g + klane * 8) * E + t * 16 + rloc] = acc[t][g];
    __syncthreads();

    // One thread per token: iterative top-8 (ties -> lowest index, matching
    // lax.top_k), then softmax over the 8 kept logits, scaled by 2.5.
    if (tid < MBLOCK) {
        float* lrow = smemf + tid * E;
        int   idxs[TOPK];
        float vals[TOPK];
#pragma unroll
        for (int s = 0; s < TOPK; ++s) {
            float best = -__builtin_inff();
            int bi = 0;
            for (int e = 0; e < E; ++e) {
                float v = lrow[e];
                if (v > best) { best = v; bi = e; }
            }
            idxs[s] = bi;
            vals[s] = best;
            lrow[bi] = -__builtin_inff();
        }
        float m = vals[0];
        float ee[TOPK], sum = 0.0f;
#pragma unroll
        for (int s = 0; s < TOPK; ++s) { ee[s] = expf(vals[s] - m); sum += ee[s]; }
        float scale = 2.5f / sum;
        size_t tok = (size_t)block0 + tid;
#pragma unroll
        for (int s = 0; s < TOPK; ++s) {
            outIdx[tok * TOPK + s] = (float)idxs[s];
            outW[tok * TOPK + s]   = ee[s] * scale;
        }
    }
}

extern "C" void kernel_launch(void* const* d_in, const int* in_sizes, int n_in,
                              void* d_out, int out_size, void* d_ws, size_t ws_size,
                              hipStream_t stream) {
    const float* X = (const float*)d_in[0];  // [4,4096,4096] fp32
    const float* W = (const float*)d_in[1];  // [256,4096] fp32
    const int T = in_sizes[0] / H;           // 16384 tokens

    unsigned* wsHi = (unsigned*)d_ws;                        // 2 MB
    unsigned* wsLo = wsHi + (size_t)NKC * NTILE * FRAG_DW;   // 2 MB

    float* outIdx = (float*)d_out;
    float* outW   = outIdx + (size_t)T * TOPK;

    // 2048 fragments x 256 dwords, one dword per thread.
    pack_w_kernel<<<(NKC * NTILE * FRAG_DW) / 256, 256, 0, stream>>>(W, wsHi, wsLo);

    gate_kernel<<<T / MBLOCK, 256, 0, stream>>>(
        X, (const uint4*)wsHi, (const uint4*)wsLo, outIdx, outW);
}
